// Net_50568944943110
// MI455X (gfx1250) — 16-bit and 8-bit matrix operand order repaired and checked by reading, compiled, not run on hardware
//
#include <hip/hip_runtime.h>
#include <hip/hip_bf16.h>

typedef __attribute__((ext_vector_type(16))) _Float16 v16h;
typedef __attribute__((ext_vector_type(8)))  float    v8f;
typedef __attribute__((ext_vector_type(16))) int      v16i;

#define NN0   5000
#define NEDGE 160000
#define HID   64
#define HEADS 2
#define FPH   32
#define OUTC  500

__global__ void k_build_edges(const int* __restrict__ ei, unsigned char* adj, int n, int e) {
  int j = blockIdx.x * blockDim.x + threadIdx.x;
  if (j < e) {
    int s = ei[j], d = ei[e + j];
    if (s >= 0 && s < n && d >= 0 && d < n) adj[(size_t)s * n + d] = 1;
  }
  if (j < n) adj[(size_t)j * n + j] = 1;
}

template <int A_IS_U8>
__global__ void k_gemm_wmma(const void* __restrict__ Ap, const float* __restrict__ B,
                            float* __restrict__ C, int M, int N, int K) {
  __shared__ _Float16 sA[32 * 32];
  __shared__ _Float16 sB[32 * 32];
  const float*         Af = (const float*)Ap;
  const unsigned char* Au = (const unsigned char*)Ap;
  int tid  = threadIdx.x;
  int bm   = blockIdx.y * 32, bn = blockIdx.x * 32;
  int lane = tid & 31, wave = tid >> 5;
  int wy = wave >> 1, wx = wave & 1;
  int hi = lane >> 4, mn = lane & 15;
  v8f acc = {};
  for (int k0 = 0; k0 < K; k0 += 32) {
#pragma unroll
    for (int p = 0; p < 8; p++) {
      int idx = tid + p * 128;
      int r = idx >> 5, kk = idx & 31;
      int gm = bm + r, gk = k0 + kk;
      int gmc = gm < M ? gm : M - 1, gkc = gk < K ? gk : K - 1;
      float va = A_IS_U8 ? (float)Au[(size_t)gmc * K + gkc] : Af[(size_t)gmc * K + gkc];
      sA[r * 32 + kk] = (_Float16)((gm < M && gk < K) ? va : 0.f);
      int gk2 = k0 + r, gn = bn + kk;
      int gk2c = gk2 < K ? gk2 : K - 1, gnc = gn < N ? gn : N - 1;
      float vb = B[(size_t)gk2c * N + gnc];
      sB[r * 32 + kk] = (_Float16)((gk2 < K && gn < N) ? vb : 0.f);
    }
    __syncthreads();
    v16h a = {}, b = {};
#pragma unroll
    for (int e = 0; e < 16; e++) {
      int i  = e >> 1;
      int ka = 16 * (i >> 2) + hi * 8 + 2 * (i & 3) + (e & 1);
      a[e]   = sA[(wy * 16 + mn) * 32 + ka];
      int kb = 16 * (e >> 3) + hi * 8 + (e & 7);
      b[e]   = sB[kb * 32 + (wx * 16 + mn)];
    }
    acc = __builtin_amdgcn_wmma_f32_16x16x32_f16(false, a, false, b, (short)0, acc, false, false);
    __syncthreads();
  }
#pragma unroll
  for (int r = 0; r < 8; r++) {
    int row = bm + wy * 16 + hi * 8 + r;
    int col = bn + wx * 16 + mn;
    if (row < M && col < N) C[(size_t)row * N + col] = acc[r];
  }
}

__global__ void __launch_bounds__(128, 1)
k_adjmm_fp8(const unsigned char* __restrict__ A, unsigned char* __restrict__ Cout, int N) {
  __shared__ int sA4[64 * 128 / 4];
  __shared__ int sB4[64 * 128 / 4];
  int tid  = threadIdx.x;
  int bm   = blockIdx.y * 64, bn = blockIdx.x * 64;
  int lane = tid & 31, wave = tid >> 5;
  int wy = wave >> 1, wx = wave & 1;
  int hi = lane >> 4, mn = lane & 15;
  v8f acc00 = {}, acc01 = {}, acc10 = {}, acc11 = {};
  for (int k0 = 0; k0 < N; k0 += 128) {
#pragma unroll
    for (int p = 0; p < 4; p++) {
      int d4 = tid + p * 128;
      int r = d4 >> 3, kd = (d4 & 7) << 4;
      int gm = bm + r, gk = k0 + kd;
      int gmc = gm < N ? gm : N - 1, gkc = gk < N ? gk : N - 16;
      uint4 v = *(const uint4*)&A[(size_t)gmc * N + gkc];
      bool ok = (gm < N) && (gk < N);
      sA4[d4 * 4 + 0] = (int)((ok ? v.x : 0u) * 0x38u);
      sA4[d4 * 4 + 1] = (int)((ok ? v.y : 0u) * 0x38u);
      sA4[d4 * 4 + 2] = (int)((ok ? v.z : 0u) * 0x38u);
      sA4[d4 * 4 + 3] = (int)((ok ? v.w : 0u) * 0x38u);
    }
#pragma unroll
    for (int p = 0; p < 4; p++) {
      int bid = tid + p * 128;
      int bc = bid & 15, bk = bid >> 4;
      int gc = bn + (bc << 2), gk = k0 + (bk << 2);
      unsigned int L[4];
#pragma unroll
      for (int r = 0; r < 4; r++) {
        int gkr  = gk + r;
        int gkrc = gkr < N ? gkr : N - 1;
        int gcc  = gc < N ? gc : N - 4;
        unsigned int v = *(const unsigned int*)&A[(size_t)gkrc * N + gcc];
        L[r] = (gkr < N && gc < N) ? v : 0u;
      }
#pragma unroll
      for (int j = 0; j < 4; j++) {
        unsigned int sel = (unsigned)(((4 + j) << 8) | j);
        unsigned int x = __builtin_amdgcn_perm(L[1], L[0], sel);
        unsigned int y = __builtin_amdgcn_perm(L[3], L[2], sel);
        unsigned int t = __builtin_amdgcn_perm(y, x, 0x05040100u);
        sB4[(bc * 4 + j) * 32 + bk] = (int)(t * 0x38u);
      }
    }
    __syncthreads();
    v16i b0 = {}, b1 = {};
#pragma unroll
    for (int i = 0; i < 16; i++) {
      int kb = 16 * (i >> 1) + hi * 8 + 4 * (i & 1);
      b0[i]  = sB4[((wx * 32 + mn) * 128 + kb) >> 2];
      b1[i]  = sB4[((wx * 32 + 16 + mn) * 128 + kb) >> 2];
    }
    {
      v16i a = {};
#pragma unroll
      for (int i = 0; i < 16; i++) {
        int ka = 64 * (i >> 3) + 32 * ((i >> 2) & 1) + 16 * ((i >> 1) & 1) + hi * 8 + 4 * (i & 1);
        a[i]   = sA4[((wy * 32 + mn) * 128 + ka) >> 2];
      }
      acc00 = __builtin_amdgcn_wmma_f32_16x16x128_fp8_fp8(a, b0, (short)0, acc00, false, false);
      acc01 = __builtin_amdgcn_wmma_f32_16x16x128_fp8_fp8(a, b1, (short)0, acc01, false, false);
    }
    {
      v16i a = {};
#pragma unroll
      for (int i = 0; i < 16; i++) {
        int ka = 64 * (i >> 3) + 32 * ((i >> 2) & 1) + 16 * ((i >> 1) & 1) + hi * 8 + 4 * (i & 1);
        a[i]   = sA4[((wy * 32 + 16 + mn) * 128 + ka) >> 2];
      }
      acc10 = __builtin_amdgcn_wmma_f32_16x16x128_fp8_fp8(a, b0, (short)0, acc10, false, false);
      acc11 = __builtin_amdgcn_wmma_f32_16x16x128_fp8_fp8(a, b1, (short)0, acc11, false, false);
    }
    __syncthreads();
  }
#pragma unroll
  for (int r = 0; r < 8; r++) {
    int r0 = bm + wy * 32 + hi * 8 + r;
    int r1 = r0 + 16;
    int c0 = bn + wx * 32 + mn;
    int c1 = c0 + 16;
    if (r0 < N && c0 < N) Cout[(size_t)r0 * N + c0] = acc00[r] > 0.5f ? 1 : 0;
    if (r0 < N && c1 < N) Cout[(size_t)r0 * N + c1] = acc01[r] > 0.5f ? 1 : 0;
    if (r1 < N && c0 < N) Cout[(size_t)r1 * N + c0] = acc10[r] > 0.5f ? 1 : 0;
    if (r1 < N && c1 < N) Cout[(size_t)r1 * N + c1] = acc11[r] > 0.5f ? 1 : 0;
  }
}

__global__ void k_scores(const float* __restrict__ h, const float* __restrict__ asrc,
                         const float* __restrict__ adst, float* ssrc, float* sdst, int n) {
  int idx = blockIdx.x * blockDim.x + threadIdx.x;
  if (idx >= n * HEADS) return;
  int hh = idx / n, i = idx % n;
  float s1 = 0.f, s2 = 0.f;
#pragma unroll
  for (int f = 0; f < FPH; f++) {
    float hv = h[(size_t)i * HID + hh * FPH + f];
    s1 += hv * asrc[hh * FPH + f];
    s2 += hv * adst[hh * FPH + f];
  }
  ssrc[hh * n + i] = s1;
  sdst[hh * n + i] = s2;
}

__global__ void k_att(const unsigned char* __restrict__ adj, const float* __restrict__ h,
                      const float* __restrict__ ssrc, const float* __restrict__ sdst,
                      float* __restrict__ outA, int n) {
  const int B = 256;
  __shared__ float red[B];
  __shared__ float sacc[FPH][B];
  int i = blockIdx.x, hh = blockIdx.y, tid = threadIdx.x;
  const unsigned int* arow4 = (const unsigned int*)(adj + (size_t)i * n);
  int nw = n >> 2;
  float si        = ssrc[hh * n + i];
  const float* sd = sdst + (size_t)hh * n;
  float mloc = -3.0e38f;
  for (int jw = tid; jw < nw; jw += B) {
    unsigned int m4 = arow4[jw];
    if (m4) {
      int j = jw << 2;
#pragma unroll
      for (int t = 0; t < 4; t++)
        if ((m4 >> (8 * t)) & 0xffu) {
          float l = si + sd[j + t];
          l       = l > 0.f ? l : 0.2f * l;
          mloc    = fmaxf(mloc, l);
        }
    }
  }
  red[tid] = mloc;
  __syncthreads();
  for (int s = B / 2; s > 0; s >>= 1) {
    if (tid < s) red[tid] = fmaxf(red[tid], red[tid + s]);
    __syncthreads();
  }
  float mx = red[0];
  __syncthreads();
  float acc[FPH];
#pragma unroll
  for (int f = 0; f < FPH; f++) acc[f] = 0.f;
  float ssum = 0.f;
  for (int jw = tid; jw < nw; jw += B) {
    unsigned int m4 = arow4[jw];
    if (m4) {
      int j = jw << 2;
#pragma unroll
      for (int t = 0; t < 4; t++)
        if ((m4 >> (8 * t)) & 0xffu) {
          float l = si + sd[j + t];
          l       = l > 0.f ? l : 0.2f * l;
          float w = __expf(l - mx);
          ssum += w;
          const float* hj = h + (size_t)(j + t) * HID + hh * FPH;
#pragma unroll
          for (int f = 0; f < FPH; f++) acc[f] += w * hj[f];
        }
    }
  }
  red[tid] = ssum;
  __syncthreads();
  for (int s = B / 2; s > 0; s >>= 1) {
    if (tid < s) red[tid] += red[tid + s];
    __syncthreads();
  }
  ssum = red[0];
#pragma unroll
  for (int f = 0; f < FPH; f++) sacc[f][tid] = acc[f];
  __syncthreads();
  if (tid < FPH) {
    float t = 0.f;
    for (int u = 0; u < B; u++) t += sacc[tid][u];
    outA[(size_t)i * HID + hh * FPH + tid] = t / ssum;
  }
}

__global__ void k_pool_score(const float* __restrict__ attn, const float* __restrict__ w,
                             float* score, int n) {
  int i = blockIdx.x * blockDim.x + threadIdx.x;
  if (i >= n) return;
  float nrm = 0.f, dot = 0.f;
#pragma unroll
  for (int f = 0; f < HID; f++) {
    float wv = w[f];
    nrm += wv * wv;
    dot += attn[(size_t)i * HID + f] * wv;
  }
  score[i] = tanhf(dot * rsqrtf(nrm));
}

__global__ void k_rank(const float* __restrict__ score, int* rank, int n) {
  int i = blockIdx.x * blockDim.x + threadIdx.x;
  if (i >= n) return;
  float s = score[i];
  int   r = 0;
  for (int j = 0; j < n; j++) {
    float t = score[j];
    r += (t > s) || (t == s && j < i);
  }
  rank[i] = r;
}

__global__ void k_perm_scatter(const int* __restrict__ rank, int* perm, int n, int k) {
  int i = blockIdx.x * blockDim.x + threadIdx.x;
  if (i >= n) return;
  int r = rank[i];
  if (r < k) perm[r] = i;
}

__global__ void k_pool_gather_f(const float* __restrict__ attn, const float* __restrict__ score,
                                const int* __restrict__ perm, float* fout, int k) {
  int idx = blockIdx.x * blockDim.x + threadIdx.x;
  if (idx >= k * HID) return;
  int r = idx / HID, c = idx % HID;
  int p   = perm[r];
  float v = attn[(size_t)p * HID + c];
  v       = v > 0.f ? v : 0.01f * v;
  fout[idx] = v * score[p];
}

__global__ void k_pool_gather_adj(const unsigned char* __restrict__ adj,
                                  const int* __restrict__ perm, unsigned char* out, int n, int k) {
  size_t idx = (size_t)blockIdx.x * blockDim.x + threadIdx.x;
  if (idx >= (size_t)k * k) return;
  int r = (int)(idx / k), c = (int)(idx % k);
  out[idx] = adj[(size_t)perm[r] * n + perm[c]];
}

__global__ void k_deg(const unsigned char* __restrict__ adj, float* deg, int n) {
  int i = blockIdx.x * blockDim.x + threadIdx.x;
  if (i >= n) return;
  const unsigned int* row4 = (const unsigned int*)(adj + (size_t)i * n);
  int s = 0, nw = n >> 2;
  for (int j = 0; j < nw; j++) s += __builtin_popcount(row4[j]);
  deg[i] = s > 0 ? (float)s : 1.0f;
}

__global__ void k_div_deg(float* agg, const float* __restrict__ deg, int n) {
  int idx = blockIdx.x * blockDim.x + threadIdx.x;
  if (idx >= n * HID) return;
  agg[idx] /= deg[idx / HID];
}

__global__ void k_scatter_up(const float* __restrict__ z, const int* __restrict__ perm, float* up,
                             int k, int c) {
  int idx = blockIdx.x * blockDim.x + threadIdx.x;
  if (idx >= k * c) return;
  int r = idx / c, cc = idx % c;
  up[(size_t)perm[r] * c + cc] = z[idx];
}

__global__ void k_combine(const float* __restrict__ t1, const float* __restrict__ t2,
                          const float* __restrict__ b, float* z, int n, int c, int relu) {
  int idx = blockIdx.x * blockDim.x + threadIdx.x;
  if (idx >= n * c) return;
  float v = t1[idx] + t2[idx] + b[idx % c];
  if (relu) v = v > 0.f ? v : 0.f;
  z[idx] = v;
}

__global__ void k_copy_f32(const float* __restrict__ s, float* d, int n) {
  int i = blockIdx.x * blockDim.x + threadIdx.x;
  if (i < n) d[i] = s[i];
}
__global__ void k_u8_f32(const unsigned char* __restrict__ s, float* d, int n) {
  int i = blockIdx.x * blockDim.x + threadIdx.x;
  if (i < n) d[i] = (float)s[i];
}
__global__ void k_zero_f32(float* d, int n) {
  int i = blockIdx.x * blockDim.x + threadIdx.x;
  if (i < n) d[i] = 0.f;
}

extern "C" void kernel_launch(void* const* d_in, const int* in_sizes, int n_in, void* d_out,
                              int out_size, void* d_ws, size_t ws_size, hipStream_t stream) {
  (void)in_sizes; (void)n_in; (void)out_size; (void)ws_size;
  const float* x  = (const float*)d_in[0];
  const int*   ei = (const int*)d_in[1];
  const float* gat_w[3]   = {(const float*)d_in[3],  (const float*)d_in[7],  (const float*)d_in[11]};
  const float* gat_as[3]  = {(const float*)d_in[4],  (const float*)d_in[8],  (const float*)d_in[12]};
  const float* gat_ad[3]  = {(const float*)d_in[5],  (const float*)d_in[9],  (const float*)d_in[13]};
  const float* pool_w[3]  = {(const float*)d_in[6],  (const float*)d_in[10], (const float*)d_in[14]};
  const float* sage_wl[3] = {(const float*)d_in[15], (const float*)d_in[18], (const float*)d_in[21]};
  const float* sage_wr[3] = {(const float*)d_in[16], (const float*)d_in[19], (const float*)d_in[22]};
  const float* sage_b[3]  = {(const float*)d_in[17], (const float*)d_in[20], (const float*)d_in[23]};

  char*  ws  = (char*)d_ws;
  size_t off = 0;
  auto   alloc = [&](size_t bytes) -> char* {
    char* p = ws + off;
    off += (bytes + 255) & ~(size_t)255;
    return p;
  };
  unsigned char* adj0  = (unsigned char*)alloc((size_t)5000 * 5000);
  unsigned char* adj1  = (unsigned char*)alloc((size_t)4000 * 4000);
  unsigned char* adj2  = (unsigned char*)alloc((size_t)3200 * 3200);
  unsigned char* adjT  = (unsigned char*)alloc((size_t)4000 * 4000);
  float* hB    = (float*)alloc((size_t)5000 * HID * 4);
  float* attnB = (float*)alloc((size_t)5000 * HID * 4);
  float* fA    = (float*)alloc((size_t)5000 * HID * 4);
  float* fB    = (float*)alloc((size_t)5000 * HID * 4);
  float* ssrc  = (float*)alloc((size_t)HEADS * 5000 * 4);
  float* sdst  = (float*)alloc((size_t)HEADS * 5000 * 4);
  float* score = (float*)alloc((size_t)5000 * 4);
  int*   rank  = (int*)alloc((size_t)5000 * 4);
  int*   perm0 = (int*)alloc((size_t)5000 * 4);
  int*   perm1 = (int*)alloc((size_t)5000 * 4);
  int*   perm2 = (int*)alloc((size_t)5000 * 4);
  float* deg   = (float*)alloc((size_t)5000 * 4);
  float* zA    = (float*)alloc((size_t)5000 * OUTC * 4);
  float* zB    = (float*)alloc((size_t)5000 * OUTC * 4);
  float* tmp1  = (float*)alloc((size_t)5000 * OUTC * 4);
  float* tmp2  = (float*)alloc((size_t)5000 * OUTC * 4);
  float* agg   = (float*)alloc((size_t)5000 * HID * 4);

  auto cdiv = [](int a, int b) { return (a + b - 1) / b; };

  hipMemsetAsync(adj0, 0, (size_t)5000 * 5000, stream);
  k_build_edges<<<cdiv(NEDGE, 256), 256, 0, stream>>>(ei, adj0, NN0, NEDGE);

  const int      ns[4]    = {5000, 4000, 3200, 2560};
  unsigned char* adjL[3]  = {adj0, adj1, adj2};
  int*           perms[3] = {perm0, perm1, perm2};
  float*         fbufs[3] = {fA, fB, fA};

  const float* fcur = x;
  int          fin  = OUTC;
  for (int lvl = 0; lvl < 3; lvl++) {
    int n = ns[lvl], k = ns[lvl + 1];
    k_gemm_wmma<0><<<dim3(cdiv(HID, 32), cdiv(n, 32)), 128, 0, stream>>>(fcur, gat_w[lvl], hB, n,
                                                                         HID, fin);
    k_scores<<<cdiv(n * HEADS, 256), 256, 0, stream>>>(hB, gat_as[lvl], gat_ad[lvl], ssrc, sdst, n);
    k_att<<<dim3(n, HEADS), 256, 0, stream>>>(adjL[lvl], hB, ssrc, sdst, attnB, n);
    k_pool_score<<<cdiv(n, 256), 256, 0, stream>>>(attnB, pool_w[lvl], score, n);
    k_rank<<<cdiv(n, 256), 256, 0, stream>>>(score, rank, n);
    k_perm_scatter<<<cdiv(n, 256), 256, 0, stream>>>(rank, perms[lvl], n, k);
    k_pool_gather_f<<<cdiv(k * HID, 256), 256, 0, stream>>>(attnB, score, perms[lvl], fbufs[lvl], k);
    k_pool_gather_adj<<<cdiv(k * k, 256), 256, 0, stream>>>(adjL[lvl], perms[lvl], adjT, n, k);
    if (lvl < 2)
      k_adjmm_fp8<<<dim3(cdiv(k, 64), cdiv(k, 64)), 128, 0, stream>>>(adjT, adjL[lvl + 1], k);
    fcur = fbufs[lvl];
    fin  = HID;
  }

  const float* zin = fA;
  for (int i = 0; i < 3; i++) {
    int            idx  = 2 - i;
    int            nUp  = ns[idx];
    int            kZ   = ns[idx + 1];
    unsigned char* A    = adjL[idx];
    int            outc = (i == 2) ? OUTC : HID;
    hipMemsetAsync(zB, 0, (size_t)nUp * HID * 4, stream);
    k_scatter_up<<<cdiv(kZ * HID, 256), 256, 0, stream>>>(zin, perms[idx], zB, kZ, HID);
    k_gemm_wmma<0><<<dim3(cdiv(outc, 32), cdiv(nUp, 32)), 128, 0, stream>>>(zB, sage_wl[i], tmp1,
                                                                            nUp, outc, HID);
    k_deg<<<cdiv(nUp, 256), 256, 0, stream>>>(A, deg, nUp);
    k_gemm_wmma<1><<<dim3(cdiv(HID, 32), cdiv(nUp, 32)), 128, 0, stream>>>(A, zB, agg, nUp, HID,
                                                                           nUp);
    k_div_deg<<<cdiv(nUp * HID, 256), 256, 0, stream>>>(agg, deg, nUp);
    k_gemm_wmma<0><<<dim3(cdiv(outc, 32), cdiv(nUp, 32)), 128, 0, stream>>>(agg, sage_wr[i], tmp2,
                                                                            nUp, outc, HID);
    k_combine<<<cdiv(nUp * outc, 256), 256, 0, stream>>>(tmp1, tmp2, sage_b[i], zA, nUp, outc,
                                                         i < 2 ? 1 : 0);
    zin = zA;
  }

  float* out = (float*)d_out;
  int    oz  = 5000 * OUTC;
  int    ox  = 2560 * HID;
  int    oa  = 2560 * 2560;
  k_copy_f32<<<cdiv(oz, 256), 256, 0, stream>>>(zA, out, oz);
  k_copy_f32<<<cdiv(ox, 256), 256, 0, stream>>>(fA, out + oz, ox);
  k_u8_f32<<<cdiv(oa, 256), 256, 0, stream>>>(adjT, out + oz + ox, oa);
  k_zero_f32<<<cdiv(2560, 256), 256, 0, stream>>>(out + oz + ox + oa, 2560);
}
